// AtomEncoder_9560597201145
// MI455X (gfx1250) — matrix operand order repaired and checked by reading, not compiled, not run on hardware
//
#include <hip/hip_runtime.h>
#include <hip/hip_bf16.h>

// ---------------------------------------------------------------------------
// AtomEncoder for MI455X (gfx1250, wave32, WMMA).
// B=32, L=512, D=256, 4D=1024, NTOKEN=128, K=6.
//
//  mlp_kernel: per-block tile of 16 positions; 8 waves; full MLP chain in LDS
//              using v_wmma_f32_16x16x32_bf16 (fp32 accum, fp32 residual path).
//  agg_kernel: bond gather/sum + emb add, [l,b,d] output.
// ---------------------------------------------------------------------------

typedef __attribute__((ext_vector_type(16))) __bf16 v16bf;
typedef __attribute__((ext_vector_type(8)))  float  v8f;

#define BATCH 32
#define SEQL  512
#define DIM   256
#define HID   1024
#define NPOS  (BATCH * SEQL)   // 16384
#define MAXB  6

__device__ __forceinline__ unsigned short f2bf(float f) {
    unsigned u = __float_as_uint(f);
    unsigned r = u + 0x7FFFu + ((u >> 16) & 1u);   // round-to-nearest-even
    return (unsigned short)(r >> 16);
}

union Frag16 { v16bf v; uint4 q[2]; };

// Load a 32-byte bf16 fragment from two 16B-aligned chunks.
__device__ __forceinline__ v16bf load_frag2(const unsigned short* c0,
                                            const unsigned short* c1) {
    Frag16 f;
    f.q[0] = *(const uint4*)c0;
    f.q[1] = *(const uint4*)c1;
    return f.v;
}

__device__ __forceinline__ v8f wmma_bf16(v16bf a, v16bf b, v8f c) {
    return __builtin_amdgcn_wmma_f32_16x16x32_bf16(
        /*neg_a=*/false, a, /*neg_b=*/false, b,
        /*c_mod=*/(short)0, c, /*reuse_a=*/false, /*reuse_b=*/false);
}

// h = relu(x @ W^T + b) :  A = xs[16][256] (LDS bf16), W row-major [1024][256]
// Wave w owns N columns [w*128, w*128+128).
__device__ __forceinline__ void expand_stage(const unsigned short* xsm,
                                             unsigned short* hsm,
                                             const unsigned short* W,
                                             const float* bias,
                                             int w, int n0, int h, int mh) {
    const v8f vzero = {0.f, 0.f, 0.f, 0.f, 0.f, 0.f, 0.f, 0.f};
    v8f acc[8];
#pragma unroll
    for (int j = 0; j < 8; ++j) acc[j] = vzero;
    const int nW = w * 128;
    for (int kk = 0; kk < 8; ++kk) {
        const int kB = kk * 32;
        const unsigned short* pa = &xsm[n0 * DIM + kB + h * 8];
        v16bf a = load_frag2(pa, pa + 16);
#pragma unroll
        for (int j = 0; j < 8; ++j) {
            const unsigned short* pb = &W[(nW + j * 16 + n0) * DIM + kB + h * 8];
            v16bf bfr = load_frag2(pb, pb + 16);
            acc[j] = wmma_bf16(a, bfr, acc[j]);
        }
    }
#pragma unroll
    for (int j = 0; j < 8; ++j) {
        const int n = nW + j * 16 + n0;
        const float bb = bias[n];
#pragma unroll
        for (int r = 0; r < 8; ++r) {
            float v = acc[j][r] + bb;
            v = v > 0.f ? v : 0.f;                    // relu
            hsm[(mh + r) * HID + n] = f2bf(v);
        }
    }
}

// x = x + (h @ W^T + b) :  A = hs[16][1024], W row-major [256][1024]
// Wave w owns N columns [w*32, w*32+32). Updates fp32 residual + bf16 copy.
__device__ __forceinline__ void project_stage(const unsigned short* hsm,
                                              unsigned short* xsm, float* xfm,
                                              const unsigned short* W,
                                              const float* bias,
                                              int w, int n0, int h, int mh) {
    const v8f vzero = {0.f, 0.f, 0.f, 0.f, 0.f, 0.f, 0.f, 0.f};
    v8f acc[2];
    acc[0] = vzero; acc[1] = vzero;
    const int nW = w * 32;
    for (int kk = 0; kk < 32; ++kk) {
        const int kB = kk * 32;
        const unsigned short* pa = &hsm[n0 * HID + kB + h * 8];
        v16bf a = load_frag2(pa, pa + 16);
#pragma unroll
        for (int j = 0; j < 2; ++j) {
            const unsigned short* pb = &W[(nW + j * 16 + n0) * HID + kB + h * 8];
            v16bf bfr = load_frag2(pb, pb + 16);
            acc[j] = wmma_bf16(a, bfr, acc[j]);
        }
    }
#pragma unroll
    for (int j = 0; j < 2; ++j) {
        const int n = nW + j * 16 + n0;
        const float bb = bias[n];
#pragma unroll
        for (int r = 0; r < 8; ++r) {
            const int m = mh + r;
            float v = xfm[m * DIM + n] + acc[j][r] + bb;   // residual in fp32
            xfm[m * DIM + n] = v;
            xsm[m * DIM + n] = f2bf(v);
        }
    }
}

// msg = x @ W5^T + b5 -> global fp32
__device__ __forceinline__ void final_stage(const unsigned short* xsm,
                                            const unsigned short* W,
                                            const float* bias,
                                            float* msgWS, int pBase,
                                            int w, int n0, int h, int mh) {
    const v8f vzero = {0.f, 0.f, 0.f, 0.f, 0.f, 0.f, 0.f, 0.f};
    v8f acc[2];
    acc[0] = vzero; acc[1] = vzero;
    const int nW = w * 32;
    for (int kk = 0; kk < 8; ++kk) {
        const int kB = kk * 32;
        const unsigned short* pa = &xsm[n0 * DIM + kB + h * 8];
        v16bf a = load_frag2(pa, pa + 16);
#pragma unroll
        for (int j = 0; j < 2; ++j) {
            const unsigned short* pb = &W[(nW + j * 16 + n0) * DIM + kB + h * 8];
            v16bf bfr = load_frag2(pb, pb + 16);
            acc[j] = wmma_bf16(a, bfr, acc[j]);
        }
    }
#pragma unroll
    for (int j = 0; j < 2; ++j) {
        const int n = nW + j * 16 + n0;
        const float bb = bias[n];
#pragma unroll
        for (int r = 0; r < 8; ++r) {
            msgWS[(size_t)(pBase + mh + r) * DIM + n] = acc[j][r] + bb;
        }
    }
}

// f32 -> bf16 weight conversion
__global__ __launch_bounds__(256) void cvt_bf16_kernel(const float* __restrict__ src,
                                                       unsigned short* __restrict__ dst,
                                                       int n) {
    int i = blockIdx.x * 256 + threadIdx.x;
    if (i < n) dst[i] = f2bf(src[i]);
}

// Fused: embedding gather-sum -> emb ws, then 5-GEMM MLP chain -> msg ws.
__global__ __launch_bounds__(256) void mlp_kernel(
    const int* __restrict__ element, const int* __restrict__ aroma,
    const int* __restrict__ charge, const int* __restrict__ segment,
    const int* __restrict__ react,
    const float* __restrict__ elem_emb, const float* __restrict__ charge_emb,
    const float* __restrict__ aroma_emb, const float* __restrict__ react_emb,
    const float* __restrict__ seg_emb, const float* __restrict__ pe,
    const unsigned short* __restrict__ w1b, const float* __restrict__ b1,
    const unsigned short* __restrict__ w2b, const float* __restrict__ b2,
    const unsigned short* __restrict__ w3b, const float* __restrict__ b3,
    const unsigned short* __restrict__ w4b, const float* __restrict__ b4,
    const unsigned short* __restrict__ w5b, const float* __restrict__ b5,
    float* __restrict__ embWS, float* __restrict__ msgWS) {
    __shared__ unsigned short xs[16 * DIM];   //  8 KB bf16 activations
    __shared__ unsigned short hs[16 * HID];   // 32 KB bf16 hidden
    __shared__ float          xf[16 * DIM];   // 16 KB fp32 residual path

    const int tid = threadIdx.x;
    const int pBase = blockIdx.x * 16;

    // --- fused embedding sum for this 16-position tile ---
    for (int i = tid; i < 16 * DIM; i += 256) {
        const int row = i >> 8;         // local position 0..15
        const int d   = i & 255;
        const int p   = pBase + row;
        const int l   = p & (SEQL - 1);
        const int e  = element[p];
        const int ar = aroma[p];
        const int ch = charge[p] + 6;
        const int sg = segment[p];
        const int rm = react[p];
        float v = elem_emb[e * DIM + d] + pe[l * DIM + d] + aroma_emb[ar * DIM + d]
                + charge_emb[ch * DIM + d] + seg_emb[sg * DIM + d]
                + react_emb[rm * DIM + d];
        embWS[(size_t)p * DIM + d] = v;
        xf[i] = v;
        xs[i] = f2bf(v);
    }
    __syncthreads();

    const int ln = tid & 31;     // lane (wave32)
    const int w  = tid >> 5;     // wave id 0..7
    const int n0 = ln & 15;      // N column / A row within subtile
    const int h  = ln >> 4;      // lane half
    const int mh = h * 8;        // C/D M offset of this lane half

    expand_stage (xs, hs, w1b, b1, w, n0, h, mh);
    __syncthreads();
    project_stage(hs, xs, xf, w2b, b2, w, n0, h, mh);
    __syncthreads();
    expand_stage (xs, hs, w3b, b3, w, n0, h, mh);
    __syncthreads();
    project_stage(hs, xs, xf, w4b, b4, w, n0, h, mh);
    __syncthreads();
    final_stage  (xs, w5b, b5, msgWS, pBase, w, n0, h, mh);
}

// out[l,b,d] = emb[p,d] + sum_k (bond[p,k] != l) * msg[b, bond[p,k], d]
__global__ __launch_bounds__(256) void agg_kernel(const int* __restrict__ bond,
                                                  const float* __restrict__ embWS,
                                                  const float* __restrict__ msgWS,
                                                  float* __restrict__ out) {
    const int g = blockIdx.x * 256 + threadIdx.x;   // 16384 * 64 lanes of float4
    const int p = g >> 6;
    const int d = (g & 63) * 4;
    const int b = p >> 9;               // L = 512
    const int l = p & (SEQL - 1);

    float4 acc = *(const float4*)&embWS[(size_t)p * DIM + d];
    const int* bp = &bond[(size_t)p * MAXB];
#pragma unroll
    for (int k = 0; k < MAXB; ++k) {
        const int bk = bp[k];
        if (bk != l) {
            const float4 m = *(const float4*)&msgWS[((size_t)b * SEQL + bk) * DIM + d];
            acc.x += m.x; acc.y += m.y; acc.z += m.z; acc.w += m.w;
        }
    }
    *(float4*)&out[((size_t)l * BATCH + b) * DIM + d] = acc;
}

extern "C" void kernel_launch(void* const* d_in, const int* in_sizes, int n_in,
                              void* d_out, int out_size, void* d_ws, size_t ws_size,
                              hipStream_t stream) {
    const int*   element    = (const int*)d_in[0];
    const int*   bond       = (const int*)d_in[1];
    const int*   aroma      = (const int*)d_in[2];
    const int*   charge     = (const int*)d_in[3];
    const int*   segment    = (const int*)d_in[4];
    const int*   react      = (const int*)d_in[5];
    const float* elem_emb   = (const float*)d_in[6];
    const float* charge_emb = (const float*)d_in[7];
    const float* aroma_emb  = (const float*)d_in[8];
    const float* react_emb  = (const float*)d_in[9];
    const float* seg_emb    = (const float*)d_in[10];
    const float* pe         = (const float*)d_in[11];
    const float* w1 = (const float*)d_in[12]; const float* b1 = (const float*)d_in[13];
    const float* w2 = (const float*)d_in[14]; const float* b2 = (const float*)d_in[15];
    const float* w3 = (const float*)d_in[16]; const float* b3 = (const float*)d_in[17];
    const float* w4 = (const float*)d_in[18]; const float* b4 = (const float*)d_in[19];
    const float* w5 = (const float*)d_in[20]; const float* b5 = (const float*)d_in[21];

    // Workspace layout (~34.3 MB):
    //   [0, 16MB)        emb  fp32  [p][d]
    //   [16MB, 32MB)     msg  fp32  [p][d]
    //   [32MB, ...)      bf16 weights w1..w5
    char* ws = (char*)d_ws;
    float* embWS = (float*)ws;
    float* msgWS = (float*)(ws + (size_t)NPOS * DIM * sizeof(float));
    unsigned short* w1b = (unsigned short*)(ws + 2 * (size_t)NPOS * DIM * sizeof(float));
    unsigned short* w2b = w1b + (size_t)HID * DIM;
    unsigned short* w3b = w2b + (size_t)DIM * HID;
    unsigned short* w4b = w3b + (size_t)HID * DIM;
    unsigned short* w5b = w4b + (size_t)DIM * HID;

    const int nW = HID * DIM;   // 262144
    const int nW5 = DIM * DIM;  // 65536
    cvt_bf16_kernel<<<(nW  + 255) / 256, 256, 0, stream>>>(w1, w1b, nW);
    cvt_bf16_kernel<<<(nW  + 255) / 256, 256, 0, stream>>>(w2, w2b, nW);
    cvt_bf16_kernel<<<(nW  + 255) / 256, 256, 0, stream>>>(w3, w3b, nW);
    cvt_bf16_kernel<<<(nW  + 255) / 256, 256, 0, stream>>>(w4, w4b, nW);
    cvt_bf16_kernel<<<(nW5 + 255) / 256, 256, 0, stream>>>(w5, w5b, nW5);

    mlp_kernel<<<NPOS / 16, 256, 0, stream>>>(
        element, aroma, charge, segment, react,
        elem_emb, charge_emb, aroma_emb, react_emb, seg_emb, pe,
        w1b, b1, w2b, b2, w3b, b3, w4b, b4, w5b, b5,
        embWS, msgWS);

    agg_kernel<<<(NPOS * (DIM / 4)) / 256, 256, 0, stream>>>(
        bond, embWS, msgWS, (float*)d_out);
}
